// QuantizedConv2d_38946763441065
// MI455X (gfx1250) — matrix operand order repaired and checked by reading, not compiled, not run on hardware
//
#include <hip/hip_runtime.h>

typedef __attribute__((ext_vector_type(16))) _Float16 v16h;
typedef __attribute__((ext_vector_type(8)))  float    v8f;
typedef unsigned int __attribute__((ext_vector_type(4))) uv4;
typedef int __attribute__((ext_vector_type(8))) iv8;
typedef int __attribute__((ext_vector_type(4))) iv4;

#define C_IN   256
#define C_OUT  256
#define WDIM   56
#define HW     3136     // 56*56
#define KTOT   2304     // 9*256  (tap-major: k = tap*256 + ci)
#define BM     128
#define BN     128
#define BK     32
#define KITERS 72       // 2304/32 ; tap = ks>>3, ci base = (ks&7)*32
#define PXTOT  100352   // 32*3136
#define NIMG   32

// ---- LDS geometry -----------------------------------------------------------
// A region: TDM-written linear tile, 128 rows x 32 halves (64B) + 16B pad per
// row (TDM pad_interval=16DW, pad_amount=4DW) -> 80B row stride, conflict-free.
#define A_ROWH   40      // halves per A row (64B data + 16B pad)
#define A_REGION 5120    // 128 * 40 halves
// B region: permuted lane-slot layout, 24-half slots + 16B skew on upper group.
#define LSTR     24
#define TILE_H   768     // 32 slots * 24 halves
#define B_REGION 6144    // 8 tiles
#define BUFH     11264   // A_REGION + B_REGION (halves)
// total LDS: 2 * 11264 halves = 45056 B

__device__ __forceinline__ int slot(int ls) {        // halves
  return ls * LSTR + ((ls & 16) >> 1);               // +8 halves if ls>=16
}

// ---------------------------------------------------------------------------
// Weight convert + reorder: OIHW int32 -> f16 [O][tap][ci] (tap-major K).
// Values 0..126 are exact in f16; dequant scale applied in the epilogue.
// ---------------------------------------------------------------------------
__global__ __launch_bounds__(256)
void wq_reorder(const int* __restrict__ wq, _Float16* __restrict__ wf) {
  const int ci = threadIdx.x;      // 0..255
  const int t  = blockIdx.x;       // 0..8
  const int o  = blockIdx.y;       // 0..255
  wf[o * KTOT + t * C_IN + ci] = (_Float16)(float)wq[o * KTOT + ci * 9 + t];
}

// Optional: x fp32 -> f16 streaming pre-pass (used when ws fits it).
__global__ __launch_bounds__(256)
void x_to_f16(const float* __restrict__ x, _Float16* __restrict__ xh, long n) {
  long i = ((long)blockIdx.x * 256 + threadIdx.x) * 4;
  if (i + 3 < n) {
    float4 v = *(const float4*)(x + i);
    xh[i + 0] = (_Float16)v.x;
    xh[i + 1] = (_Float16)v.y;
    xh[i + 2] = (_Float16)v.z;
    xh[i + 3] = (_Float16)v.w;
  }
}

// ---------------------------------------------------------------------------
// Implicit-GEMM conv via V_WMMA_F32_16X16X32_F16 + TDM weight staging.
//   M = Cout(256), N = pixels(100352), K = 9*256 tap-major.
//   A tile (128x64B, row stride 4608B) staged by TENSOR_LOAD_TO_LDS with LDS
//   padding; B (im2col gather) staged per-lane with one bounds check/slab.
// ---------------------------------------------------------------------------
template <typename XT>
__global__ __launch_bounds__(256)
void conv_implicit_gemm_wmma(const XT* __restrict__ x,
                             const _Float16* __restrict__ wf,
                             const float* __restrict__ bias,
                             const float* __restrict__ scale,
                             float* __restrict__ out) {
  __shared__ _Float16 smem[2 * BUFH];

  const int tid  = threadIdx.x;
  const int lane = tid & 31;
  const int wave = tid >> 5;
  const int wm   = wave & 3;   // 4 waves along M
  const int wn   = wave >> 2;  // 2 waves along N
  const int n0   = blockIdx.x * BN;  // pixel base
  const int m0   = blockIdx.y * BM;  // cout base

  // --- TDM descriptor pieces (uniform) -------------------------------------
  const unsigned long a_gbase =
      (unsigned long)(const void*)(wf + (long)m0 * KTOT);  // row m0, k=0
  const unsigned lds_base = (unsigned)(unsigned long)(const void*)&smem[0];

  iv8 g1;
  g1[0] = (1 << 16)            // data_size = 2B
        | (1 << 20)            // pad_enable
        | (3 << 22)            // pad_interval: 16 DWORDs (one 64B row)
        | (3 << 25);           // pad_amount: 4 DWORDs (16B)
  g1[1] = (KTOT & 0xffff) << 16;                      // tensor_dim0 [63:48]
  g1[2] = (KTOT >> 16) | ((BM & 0xffff) << 16);       // dim0 hi / dim1 lo
  g1[3] = (BK << 16);                                 // dim1 hi=0 / tile_dim0=32
  g1[4] = BM;                                         // tile_dim1=128, tile_dim2=0
  g1[5] = KTOT;                                       // tensor_dim0_stride lo32
  g1[6] = 0;
  g1[7] = 0;
  const iv4 gz4 = {0, 0, 0, 0};
  const iv8 gz8 = {0, 0, 0, 0, 0, 0, 0, 0};

  auto tdm_load_a = [&](int ks, int buf) {
    uv4 g0;
    unsigned long ga = a_gbase + (unsigned long)ks * (BK * 2);
    g0[0] = 1u;                                       // count=1, user mode
    g0[1] = lds_base + (unsigned)buf * (BUFH * 2);    // lds_addr (bytes)
    g0[2] = (unsigned)ga;
    g0[3] = (unsigned)((ga >> 32) & 0x1FFFFFFu) | (2u << 30);  // type=2
    __builtin_amdgcn_tensor_load_to_lds(g0, g1, gz4, gz4, gz8, 0);
  };

  // --- B (im2col x) staging: thread -> (pixel = tid>>1, k-half = (tid&1)*16)
  const int b_n  = tid >> 1;
  const int b_kh = tid & 1;
  const int b_nt = b_n >> 4, b_nn = b_n & 15;
  const int b_d  = b_nt * TILE_H + slot(b_nn + (b_kh << 4));
  const int p    = n0 + b_n;
  const int img  = p / HW;
  const int rem  = p - img * HW;
  const int oh   = rem / WDIM;
  const int ow   = rem - oh * WDIM;
  const XT* xin  = x + (long)img * (C_IN * HW) + (long)(b_kh << 4) * HW;

  v8f acc[2][4] = {};
  v16h b_r;

  auto fetch_b = [&](int ks) {
    const int t  = ks >> 3;          // tap, uniform per slab
    const int dh = t / 3 - 1;        // scalar math
    const int dw = t - (t / 3) * 3 - 1;
    const int ih = oh + dh;
    const int iw = ow + dw;
    const XT* src = xin + (long)((ks & 7) << 5) * HW + (ih * WDIM + iw);
    if (((unsigned)ih < (unsigned)WDIM) & ((unsigned)iw < (unsigned)WDIM)) {
#pragma unroll
      for (int j = 0; j < 16; ++j) b_r[j] = (_Float16)src[j * HW];
    } else {
#pragma unroll
      for (int j = 0; j < 16; ++j) b_r[j] = (_Float16)0.f;
    }
  };

  auto commit_b = [&](int buf) {
    union { v16h v; uint4 u[2]; } br;
    br.v = b_r;
    _Float16* bb = &smem[buf * BUFH + A_REGION + b_d];
    *(uint4*)(bb)     = br.u[0];
    *(uint4*)(bb + 8) = br.u[1];
  };

  // A fragment from linear padded rows (contiguous k order, same as the B slots):
  // lane L<16: M=L, halves {k0..15}; lane L+16: M=L, {k16..31}.
  const int arow = lane & 15;
  const int ahi  = (lane >> 4) << 4;   // +16 halves for upper lane group

  auto compute = [&](int buf) {
    const _Float16* As = &smem[buf * BUFH];
    const _Float16* Bs = As + A_REGION;
    const int lsl = slot(lane);
    v16h af[2], bf[4];
#pragma unroll
    for (int i = 0; i < 2; ++i) {
      const _Float16* pa = As + ((wm * 2 + i) * 16 + arow) * A_ROWH + ahi;
      union { uint4 u[2]; v16h v; } r;
      r.u[0] = *(const uint4*)pa;          // k 0..7  (or 16..23)
      r.u[1] = *(const uint4*)(pa + 8);    // k 8..15 (or 24..31)
      af[i] = r.v;
    }
#pragma unroll
    for (int j = 0; j < 4; ++j) {
      const _Float16* pb = Bs + (wn * 4 + j) * TILE_H + lsl;
      union { uint4 u[2]; v16h v; } r;
      r.u[0] = *(const uint4*)pb;
      r.u[1] = *(const uint4*)(pb + 8);
      bf[j] = r.v;
    }
#pragma unroll
    for (int i = 0; i < 2; ++i)
#pragma unroll
      for (int j = 0; j < 4; ++j)
        acc[i][j] = __builtin_amdgcn_wmma_f32_16x16x32_f16(
            false, af[i], false, bf[j], (short)0, acc[i][j], false, false);
  };

  // --- prologue: slab 0 into buffer 0 ---------------------------------------
  if (wave == 0) tdm_load_a(0, 0);     // EXEC ignored by TDM; real branch keeps
  fetch_b(0);                          // other waves from duplicating the DMA
  commit_b(0);
  __builtin_amdgcn_s_wait_tensorcnt(0);  // no-op for waves with TENSORcnt==0
  __syncthreads();

  for (int ks = 0; ks < KITERS; ++ks) {
    if (ks + 1 < KITERS) {
      if (wave == 0) tdm_load_a(ks + 1, (ks + 1) & 1);  // TDM in flight...
      fetch_b(ks + 1);                                  // gather in flight...
    }
    compute(ks & 1);                                    // ...under the WMMAs
    if (ks + 1 < KITERS) commit_b((ks + 1) & 1);
    __builtin_amdgcn_s_wait_tensorcnt(0);  // unconditional: free if cnt==0
    __syncthreads();
  }

  // Epilogue: out = scale * acc + bias. C/D layout (ISA 7.12.2):
  // VGPR r, lanes 0-15 -> M=r, N=lane ; lanes 16-31 -> M=r+8, N=lane-16.
  const float s = scale[0];
#pragma unroll
  for (int i = 0; i < 2; ++i) {
    const int co0 = m0 + (wm * 2 + i) * 16 + ((lane >> 4) << 3);
    float bv[8];
#pragma unroll
    for (int r = 0; r < 8; ++r) bv[r] = bias[co0 + r];
#pragma unroll
    for (int j = 0; j < 4; ++j) {
      const int pp = n0 + (wn * 4 + j) * 16 + (lane & 15);
      const int im = pp / HW;
      const int rm = pp - im * HW;
      float* ob = out + (long)im * (C_OUT * HW) + rm;
#pragma unroll
      for (int r = 0; r < 8; ++r)
        ob[(long)(co0 + r) * HW] = s * acc[i][j][r] + bv[r];
    }
  }
}

extern "C" void kernel_launch(void* const* d_in, const int* in_sizes, int n_in,
                              void* d_out, int out_size, void* d_ws, size_t ws_size,
                              hipStream_t stream) {
  const float* x     = (const float*)d_in[0];
  const int*   wq    = (const int*)d_in[1];
  const float* bias  = (const float*)d_in[2];
  const float* scale = (const float*)d_in[3];
  float*       out   = (float*)d_out;

  const size_t WBYTES = (size_t)C_OUT * KTOT * sizeof(_Float16);  // 1.18 MB
  const long   XELEMS = (long)NIMG * C_IN * HW;                   // 25.69 M

  _Float16* wf = (_Float16*)d_ws;
  wq_reorder<<<dim3(9, C_OUT), 256, 0, stream>>>(wq, wf);

  dim3 grid(PXTOT / BN, C_OUT / BM);  // 784 x 2

  if (ws_size >= WBYTES + (size_t)XELEMS * sizeof(_Float16)) {
    // Pre-convert x to f16: removes all v_cvt from the hot loop and halves
    // gather bytes. One extra streaming pass (~150 MB @ 23.3 TB/s ~ 7 us).
    _Float16* xh = (_Float16*)((char*)d_ws + WBYTES);
    x_to_f16<<<(unsigned)((XELEMS / 4 + 255) / 256), 256, 0, stream>>>(x, xh, XELEMS);
    conv_implicit_gemm_wmma<_Float16><<<grid, 256, 0, stream>>>(xh, wf, bias, scale, out);
  } else {
    conv_implicit_gemm_wmma<float><<<grid, 256, 0, stream>>>(x, wf, bias, scale, out);
  }
}
